// MoE_49177375539355
// MI455X (gfx1250) — matrix operand order repaired and checked by reading, not compiled, not run on hardware
//
#include <hip/hip_runtime.h>
#include <hip/hip_bf16.h>

#define N_TOK 8192
#define DIM   2048
#define HID   1408
#define NEXP  8

typedef __bf16 bf16_t;
typedef bf16_t v16bf __attribute__((ext_vector_type(16)));
typedef float  v8f   __attribute__((ext_vector_type(8)));
typedef int    v4i_t __attribute__((ext_vector_type(4)));

union Frag16 { uint4 q[2]; v16bf v; };

#ifndef __has_builtin
#define __has_builtin(x) 0
#endif

#if defined(__AMDGCN__) && __has_builtin(__builtin_amdgcn_global_load_async_to_lds_b128)
#define HAVE_ASYNC_LDS 1
#else
#define HAVE_ASYNC_LDS 0
#endif

// async b128 builtin operand types: (v4i AS(1)*, v4i AS(3)*, imm offset, imm cpol)
typedef __attribute__((address_space(1))) v4i_t* g_v4i_p;
typedef __attribute__((address_space(3))) v4i_t* l_v4i_p;

__device__ __forceinline__ void wait_asynccnt0() {
#if __has_builtin(__builtin_amdgcn_s_wait_asynccnt)
    __builtin_amdgcn_s_wait_asynccnt(0);
#else
    asm volatile("s_wait_asynccnt 0x0" ::: "memory");
#endif
}

// native bf16 convert (lowers to v_cvt_pk_bf16_f32 on gfx1250)
__device__ __forceinline__ unsigned short f2bf(float f) {
    union { __bf16 b; unsigned short u; } cv;
    cv.b = (__bf16)f;
    return cv.u;
}

// ---------------------------------------------------------------------------
// Prepass: x fp32 -> bf16 (once per call; removes conversion from hot loops)
// ---------------------------------------------------------------------------
__global__ __launch_bounds__(256)
void cvt_x_bf16(const float* __restrict__ X, unsigned short* __restrict__ Xb) {
    const long i = ((long)blockIdx.x * 256 + threadIdx.x) * 8;
    if (i >= (long)N_TOK * DIM) return;
    float4 f0 = *(const float4*)(X + i);
    float4 f1 = *(const float4*)(X + i + 4);
    alignas(16) unsigned short h[8] = {
        f2bf(f0.x), f2bf(f0.y), f2bf(f0.z), f2bf(f0.w),
        f2bf(f1.x), f2bf(f1.y), f2bf(f1.z), f2bf(f1.w) };
    *(uint4*)(Xb + i) = *(const uint4*)h;
}

// ---------------------------------------------------------------------------
// Router: sigmoid(x @ gate_w^T), top-2 (bias added for selection only),
// normalized scores scattered into dense combine[N, E]. One wave32 per token.
// ---------------------------------------------------------------------------
__global__ __launch_bounds__(256)
void router_kernel(const float* __restrict__ x,
                   const float* __restrict__ gw,
                   const float* __restrict__ bias,
                   float* __restrict__ combine) {
    const int wave = threadIdx.x >> 5;
    const int lane = threadIdx.x & 31;
    const int n = blockIdx.x * 8 + wave;
    if (n >= N_TOK) return;
    const float* xr = x + (long)n * DIM;
    float dot[NEXP];
    for (int e = 0; e < NEXP; ++e) {
        const float* wr = gw + (long)e * DIM;
        float s = 0.f;
        for (int d = lane; d < DIM; d += 32)
            s += xr[d] * wr[d];
        for (int off = 16; off; off >>= 1)
            s += __shfl_xor(s, off, 32);
        dot[e] = s;
    }
    if (lane == 0) {
        float sig[NEXP], b[NEXP];
        for (int e = 0; e < NEXP; ++e) {
            sig[e] = 1.f / (1.f + __expf(-dot[e]));
            b[e] = sig[e] + bias[e];
        }
        int i1 = 0;
        for (int e = 1; e < NEXP; ++e) if (b[e] > b[i1]) i1 = e;
        int i2 = (i1 == 0) ? 1 : 0;
        for (int e = 0; e < NEXP; ++e) if (e != i1 && b[e] > b[i2]) i2 = e;
        const float s1 = sig[i1], s2 = sig[i2];
        const float inv = 1.f / (s1 + s2 + 1e-20f);
        float* c = combine + (long)n * NEXP;
        for (int e = 0; e < NEXP; ++e) c[e] = 0.f;
        c[i1] = s1 * inv;
        c[i2] = s2 * inv;
    }
}

// ---------------------------------------------------------------------------
// GEMM1: h = silu(X @ W1^T) * (X @ W3^T), bf16 WMMA, output bf16 to ws.
// Block tile 64(M) x 128(H); 8 waves, each wave 32x32 (2x2 wmma tiles).
// ABF16: A comes pre-converted (bf16) -> async DMA global->LDS.
// ---------------------------------------------------------------------------
template <bool ABF16>
__global__ __launch_bounds__(256)
void moe_gemm_swiglu(const float* __restrict__ X,            // [N_TOK, DIM] fp32
                     const unsigned short* __restrict__ Xb,  // [N_TOK, DIM] bf16
                     const float* __restrict__ W1,           // [HID, DIM]
                     const float* __restrict__ W3,           // [HID, DIM]
                     unsigned short* __restrict__ Hb) {      // [N_TOK, HID] bf16
    __shared__ unsigned short As [64][40];
    __shared__ unsigned short B1s[128][40];
    __shared__ unsigned short B3s[128][40];

    const int t    = threadIdx.x;
    const int lane = t & 31;
    const int wave = t >> 5;
    const int wm   = wave & 1;         // 0..1  (M within block)
    const int wn   = wave >> 1;        // 0..3  (N within block)
    const int mBase = blockIdx.x * 64;
    const int nBase = blockIdx.y * 128;

    const int ar = t >> 2;             // 0..63
    const int ac = (t & 3) * 8;        // 0,8,16,24
    const int br = t >> 1;             // 0..127
    const int bc = (t & 1) * 16;       // 0,16

    const v8f vzero = {0.f,0.f,0.f,0.f,0.f,0.f,0.f,0.f};
    v8f acc1[2][2], acc3[2][2];
    for (int im = 0; im < 2; ++im)
        for (int in = 0; in < 2; ++in) { acc1[im][in] = vzero; acc3[im][in] = vzero; }

    for (int k0 = 0; k0 < DIM; k0 += 32) {
        // ---- stage A tile ----
        if (ABF16) {
#if HAVE_ASYNC_LDS
            __builtin_amdgcn_global_load_async_to_lds_b128(
                (g_v4i_p)(Xb + (long)(mBase + ar) * DIM + k0 + ac),
                (l_v4i_p)&As[ar][ac], 0, 0);
#else
            *(uint4*)&As[ar][ac] =
                *(const uint4*)(Xb + (long)(mBase + ar) * DIM + k0 + ac);
#endif
        } else {
            const float* src = X + (long)(mBase + ar) * DIM + k0 + ac;
            float4 f0 = *(const float4*)(src);
            float4 f1 = *(const float4*)(src + 4);
            alignas(16) unsigned short h[8] = {
                f2bf(f0.x), f2bf(f0.y), f2bf(f0.z), f2bf(f0.w),
                f2bf(f1.x), f2bf(f1.y), f2bf(f1.z), f2bf(f1.w) };
            *(uint4*)&As[ar][ac] = *(const uint4*)h;
        }
        // ---- stage B1/B3 tiles (fp32 -> bf16, overlaps async A copy) ----
        {
            const float* s1 = W1 + (long)(nBase + br) * DIM + k0 + bc;
            float4 a0 = *(const float4*)(s1);
            float4 a1 = *(const float4*)(s1 + 4);
            float4 a2 = *(const float4*)(s1 + 8);
            float4 a3 = *(const float4*)(s1 + 12);
            alignas(16) unsigned short h[16] = {
                f2bf(a0.x),f2bf(a0.y),f2bf(a0.z),f2bf(a0.w),
                f2bf(a1.x),f2bf(a1.y),f2bf(a1.z),f2bf(a1.w),
                f2bf(a2.x),f2bf(a2.y),f2bf(a2.z),f2bf(a2.w),
                f2bf(a3.x),f2bf(a3.y),f2bf(a3.z),f2bf(a3.w) };
            *(uint4*)&B1s[br][bc]     = *(const uint4*)(h);
            *(uint4*)&B1s[br][bc + 8] = *(const uint4*)(h + 8);
        }
        {
            const float* s3 = W3 + (long)(nBase + br) * DIM + k0 + bc;
            float4 a0 = *(const float4*)(s3);
            float4 a1 = *(const float4*)(s3 + 4);
            float4 a2 = *(const float4*)(s3 + 8);
            float4 a3 = *(const float4*)(s3 + 12);
            alignas(16) unsigned short h[16] = {
                f2bf(a0.x),f2bf(a0.y),f2bf(a0.z),f2bf(a0.w),
                f2bf(a1.x),f2bf(a1.y),f2bf(a1.z),f2bf(a1.w),
                f2bf(a2.x),f2bf(a2.y),f2bf(a2.z),f2bf(a2.w),
                f2bf(a3.x),f2bf(a3.y),f2bf(a3.z),f2bf(a3.w) };
            *(uint4*)&B3s[br][bc]     = *(const uint4*)(h);
            *(uint4*)&B3s[br][bc + 8] = *(const uint4*)(h + 8);
        }
#if HAVE_ASYNC_LDS
        if (ABF16) wait_asynccnt0();   // async writes must land before barrier
#endif
        __syncthreads();

        if (k0 + 32 < DIM) {  // prefetch next K-tile of the weights
            __builtin_prefetch(W1 + (long)(nBase + br) * DIM + k0 + 32 + bc, 0, 1);
            __builtin_prefetch(W3 + (long)(nBase + br) * DIM + k0 + 32 + bc, 0, 1);
        }

        // ---- fragments (contiguous k = 16h + s per lane-half) + WMMA ----
        Frag16 a[2], b1[2], b3[2];
        const int ksel = (lane >> 4) * 16;  // A: K half per lane-half
        for (int im = 0; im < 2; ++im) {
            const int row = wm * 32 + im * 16 + (lane & 15);
            a[im].q[0] = *(const uint4*)&As[row][ksel];
            a[im].q[1] = *(const uint4*)&As[row][ksel + 8];
        }
        const int kb = (lane >> 4) * 16;    // B: K half per lane-half
        for (int in = 0; in < 2; ++in) {
            const int row = wn * 32 + in * 16 + (lane & 15);
            b1[in].q[0] = *(const uint4*)&B1s[row][kb];
            b1[in].q[1] = *(const uint4*)&B1s[row][kb + 8];
            b3[in].q[0] = *(const uint4*)&B3s[row][kb];
            b3[in].q[1] = *(const uint4*)&B3s[row][kb + 8];
        }
        for (int im = 0; im < 2; ++im)
            for (int in = 0; in < 2; ++in) {
                acc1[im][in] = __builtin_amdgcn_wmma_f32_16x16x32_bf16(
                    false, a[im].v, false, b1[in].v, (short)0, acc1[im][in], false, false);
                acc3[im][in] = __builtin_amdgcn_wmma_f32_16x16x32_bf16(
                    false, a[im].v, false, b3[in].v, (short)0, acc3[im][in], false, false);
            }
        __syncthreads();
    }

    // fused SwiGLU epilogue -> bf16 h
    for (int im = 0; im < 2; ++im)
        for (int in = 0; in < 2; ++in)
            for (int v = 0; v < 8; ++v) {
                const float h1 = acc1[im][in][v];
                const float h3 = acc3[im][in][v];
                const float val = (h1 / (1.f + __expf(-h1))) * h3;
                const int row = mBase + wm * 32 + im * 16 + (lane >> 4) * 8 + v;
                const int col = nBase + wn * 32 + in * 16 + (lane & 15);
                Hb[(long)row * HID + col] = f2bf(val);
            }
}

// ---------------------------------------------------------------------------
// GEMM2: out (+)= combine[:,e] * (h @ W2^T). A is bf16 ws (async-copied to
// LDS via GLOBAL_LOAD_ASYNC_TO_LDS_B128 when available), B fp32 -> bf16.
// ---------------------------------------------------------------------------
__global__ __launch_bounds__(256)
void moe_gemm_down(const unsigned short* __restrict__ Hb, // [N_TOK, HID] bf16
                   const float* __restrict__ W2,          // [DIM, HID]
                   float* __restrict__ Out,               // [N_TOK, DIM]
                   const float* __restrict__ combine_col, // stride NEXP; null => 1.0
                   int accumulate) {
    __shared__ unsigned short As[64][40];
    __shared__ unsigned short Bs[128][40];

    const int t    = threadIdx.x;
    const int lane = t & 31;
    const int wave = t >> 5;
    const int wm   = wave & 1;
    const int wn   = wave >> 1;
    const int mBase = blockIdx.x * 64;
    const int nBase = blockIdx.y * 128;

    const int ar = t >> 2;
    const int ac = (t & 3) * 8;
    const int br = t >> 1;
    const int bc = (t & 1) * 16;

    const v8f vzero = {0.f,0.f,0.f,0.f,0.f,0.f,0.f,0.f};
    v8f acc[2][2];
    for (int im = 0; im < 2; ++im)
        for (int in = 0; in < 2; ++in) acc[im][in] = vzero;

    for (int k0 = 0; k0 < HID; k0 += 32) {
        // ---- A tile: already bf16 -> async DMA global->LDS (ASYNCcnt path)
#if HAVE_ASYNC_LDS
        __builtin_amdgcn_global_load_async_to_lds_b128(
            (g_v4i_p)(Hb + (long)(mBase + ar) * HID + k0 + ac),
            (l_v4i_p)&As[ar][ac], 0, 0);
#else
        *(uint4*)&As[ar][ac] =
            *(const uint4*)(Hb + (long)(mBase + ar) * HID + k0 + ac);
#endif
        // ---- B tile: fp32 -> bf16 (overlaps with async A copy) ----
        {
            const float* sb = W2 + (long)(nBase + br) * HID + k0 + bc;
            float4 a0 = *(const float4*)(sb);
            float4 a1 = *(const float4*)(sb + 4);
            float4 a2 = *(const float4*)(sb + 8);
            float4 a3 = *(const float4*)(sb + 12);
            alignas(16) unsigned short h[16] = {
                f2bf(a0.x),f2bf(a0.y),f2bf(a0.z),f2bf(a0.w),
                f2bf(a1.x),f2bf(a1.y),f2bf(a1.z),f2bf(a1.w),
                f2bf(a2.x),f2bf(a2.y),f2bf(a2.z),f2bf(a2.w),
                f2bf(a3.x),f2bf(a3.y),f2bf(a3.z),f2bf(a3.w) };
            *(uint4*)&Bs[br][bc]     = *(const uint4*)(h);
            *(uint4*)&Bs[br][bc + 8] = *(const uint4*)(h + 8);
        }
#if HAVE_ASYNC_LDS
        wait_asynccnt0();   // our async writes must land before the barrier
#endif
        __syncthreads();

        if (k0 + 32 < HID) {
            __builtin_prefetch(W2 + (long)(nBase + br) * HID + k0 + 32 + bc, 0, 1);
        }

        Frag16 a[2], b[2];
        const int ksel = (lane >> 4) * 16;
        for (int im = 0; im < 2; ++im) {
            const int row = wm * 32 + im * 16 + (lane & 15);
            a[im].q[0] = *(const uint4*)&As[row][ksel];
            a[im].q[1] = *(const uint4*)&As[row][ksel + 8];
        }
        const int kb = (lane >> 4) * 16;
        for (int in = 0; in < 2; ++in) {
            const int row = wn * 32 + in * 16 + (lane & 15);
            b[in].q[0] = *(const uint4*)&Bs[row][kb];
            b[in].q[1] = *(const uint4*)&Bs[row][kb + 8];
        }
        for (int im = 0; im < 2; ++im)
            for (int in = 0; in < 2; ++in)
                acc[im][in] = __builtin_amdgcn_wmma_f32_16x16x32_bf16(
                    false, a[im].v, false, b[in].v, (short)0, acc[im][in], false, false);
        __syncthreads();
    }

    for (int im = 0; im < 2; ++im)
        for (int in = 0; in < 2; ++in)
            for (int v = 0; v < 8; ++v) {
                const int row = mBase + wm * 32 + im * 16 + (lane >> 4) * 8 + v;
                const int col = nBase + wn * 32 + in * 16 + (lane & 15);
                const float s = combine_col ? combine_col[(long)row * NEXP] : 1.0f;
                const float val = s * acc[im][in][v];
                const long idx = (long)row * DIM + col;
                if (accumulate) Out[idx] += val;
                else            Out[idx]  = val;
            }
}

// ---------------------------------------------------------------------------
extern "C" void kernel_launch(void* const* d_in, const int* in_sizes, int n_in,
                              void* d_out, int out_size, void* d_ws, size_t ws_size,
                              hipStream_t stream) {
    const float* x    = (const float*)d_in[0];
    const float* gw   = (const float*)d_in[1];
    const float* bias = (const float*)d_in[2];
    const float* w1   = (const float*)d_in[3];
    const float* w2   = (const float*)d_in[4];
    const float* w3   = (const float*)d_in[5];
    const float* sw1  = (const float*)d_in[6];
    const float* sw2  = (const float*)d_in[7];
    const float* sw3  = (const float*)d_in[8];
    float* out = (float*)d_out;

    // workspace layout
    const size_t hbytes = (size_t)N_TOK * HID * 2;   // h bf16
    const size_t cbytes = (size_t)N_TOK * NEXP * 4;  // combine f32
    const size_t xbytes = (size_t)N_TOK * DIM * 2;   // x bf16 (optional)
    unsigned short* hbuf = (unsigned short*)d_ws;
    float* combine = (float*)((char*)d_ws + hbytes);
    unsigned short* xb = (unsigned short*)((char*)d_ws + hbytes + cbytes);
    const bool use_xb = ws_size >= hbytes + cbytes + xbytes;

    router_kernel<<<N_TOK / 8, 256, 0, stream>>>(x, gw, bias, combine);
    if (use_xb) {
        const long nelem = (long)N_TOK * DIM;
        cvt_x_bf16<<<(int)(nelem / (256 * 8)), 256, 0, stream>>>(x, xb);
    }

    const dim3 g1(N_TOK / 64, HID / 128);   // 128 x 11
    const dim3 g2(N_TOK / 64, DIM / 128);   // 128 x 16

    // shared expert first: overwrite mode (no zero-init / atomics needed)
    if (use_xb)
        moe_gemm_swiglu<true ><<<g1, 256, 0, stream>>>(x, xb, sw1, sw3, hbuf);
    else
        moe_gemm_swiglu<false><<<g1, 256, 0, stream>>>(x, nullptr, sw1, sw3, hbuf);
    moe_gemm_down<<<g2, 256, 0, stream>>>(hbuf, sw2, out, nullptr, 0);

    // routed experts: accumulate, masked by combine column e
    for (int e = 0; e < NEXP; ++e) {
        const float* e1 = w1 + (size_t)e * HID * DIM;
        const float* e3 = w3 + (size_t)e * HID * DIM;
        if (use_xb)
            moe_gemm_swiglu<true ><<<g1, 256, 0, stream>>>(x, xb, e1, e3, hbuf);
        else
            moe_gemm_swiglu<false><<<g1, 256, 0, stream>>>(x, nullptr, e1, e3, hbuf);
        moe_gemm_down<<<g2, 256, 0, stream>>>(
            hbuf, w2 + (size_t)e * DIM * HID, out, combine + e, 1);
    }
}
